// SpatioTemporalGNN_20899310862960
// MI455X (gfx1250) — matrix operand order repaired and checked by reading, compiled, not run on hardware
//
#include <hip/hip_runtime.h>

typedef __attribute__((ext_vector_type(16))) _Float16 v16h;
typedef __attribute__((ext_vector_type(8)))  _Float16 v8h;
typedef __attribute__((ext_vector_type(8)))  float    v8f;

#define B_    4
#define T_    16
#define N_    1000
#define FIN   8
#define E_    8000
#define EP    9000      // edges + self loops
#define HID   64
#define HEADS 4
#define LAY   5
#define G_    64        // B*T graphs
#define ROWS  64000     // G_*N_
#define HC    256       // HEADS*HID
#define KGRU  64000     // N_*HID
#define GRUO  192       // 3*HID
#define KCH   100       // K-split chunks for GRU GEMM
#define KCW   20        // k32 steps per chunk (100*20*32 = 64000)

// ---------------- weight preparation ----------------

// Pack Wg[l] (64x256 f32) into WMMA B-fragment order:
// WgP[l][ntile(16)][kc(2)][lane(32)][e(16)] f16 -> one contiguous v16h per lane.
__global__ void k_pack_wg(const float* __restrict__ Wg, _Float16* __restrict__ WgP) {
    int i = blockIdx.x * 256 + threadIdx.x;
    if (i >= LAY * 16 * 2 * 32 * 16) return;
    int e    = i & 15;
    int lane = (i >> 4) & 31;
    int kc   = (i >> 9) & 1;
    int nt   = (i >> 10) & 15;
    int l    = i >> 14;
    int col  = nt * 16 + (lane & 15);
    int k    = kc * 32 + ((e < 8) ? 0 : 16) + ((lane < 16) ? 0 : 8) + (e & 7);
    WgP[i] = (_Float16)Wg[(l * 64 + k) * HC + col];
}

__global__ void k_conv_f16(const float* __restrict__ src, _Float16* __restrict__ dst, int n) {
    for (int i = blockIdx.x * blockDim.x + threadIdx.x; i < n; i += gridDim.x * blockDim.x)
        dst[i] = (_Float16)src[i];
}

// ---------------- deterministic CSR build (edges identical for all graphs) ----

__global__ void k_zero_int(int* p, int n) {
    int i = blockIdx.x * blockDim.x + threadIdx.x;
    if (i < n) p[i] = 0;
}

__global__ void k_edges(const int* __restrict__ ei, int* __restrict__ dstA,
                        int* __restrict__ srcA, int* __restrict__ deg) {
    int e = blockIdx.x * blockDim.x + threadIdx.x;
    if (e >= EP) return;
    int s, d;
    if (e < E_) { s = ei[e]; d = ei[E_ + e]; }
    else        { s = d = e - E_; }               // self loop
    srcA[e] = s; dstA[e] = d;
    atomicAdd(&deg[d], 1);                        // integer: deterministic
}

__global__ void k_scan(const int* __restrict__ deg, int* __restrict__ start) {
    if (blockIdx.x == 0 && threadIdx.x == 0) {
        int acc = 0;
        for (int n = 0; n < N_; n++) { start[n] = acc; acc += deg[n]; }
        start[N_] = acc;
    }
}

// stable counting-sort scatter: rank = #earlier edges with same dst (deterministic)
__global__ void k_scatter(const int* __restrict__ dstA, const int* __restrict__ srcA,
                          const int* __restrict__ start, int* __restrict__ csrc) {
    int e = blockIdx.x * blockDim.x + threadIdx.x;
    if (e >= EP) return;
    int d = dstA[e];
    int r = 0;
    for (int e2 = 0; e2 < e; e2++) r += (dstA[e2] == d);
    csrc[start[d] + r] = srcA[e];
}

// ---------------- input projection [ROWS,8] @ [8,64] ----------------

__global__ void k_infc(const float* __restrict__ xs, const float* __restrict__ Win,
                       const float* __restrict__ bin, _Float16* __restrict__ Xh) {
    int i = blockIdx.x * blockDim.x + threadIdx.x;
    if (i >= ROWS * HID) return;
    int c = i & 63;
    int r = i >> 6;
    const float* xp = xs + r * FIN;
    float v = bin[c];
#pragma unroll
    for (int f = 0; f < FIN; f++) v += xp[f] * Win[f * HID + c];
    Xh[i] = (_Float16)v;
}

// ---------------- GAT feature GEMM: H = Xh @ Wg[l]  (WMMA f16->f32) -----------
// Block stages the layer's 32KB packed weights in LDS via async-to-LDS DMA.
// One wave = 16-row strip x 8 ntiles: A fragments stay in registers (8x reuse).

__global__ void __launch_bounds__(256) k_gemm_gat(const _Float16* __restrict__ Xh,
                                                  const _Float16* __restrict__ WgP,
                                                  _Float16* __restrict__ H) {
    __shared__ __align__(32) _Float16 sw[16 * 2 * 32 * 16];   // 32 KB
    {
        // cooperative async copy: 256 threads x 8 x 16B = 32KB (ASYNCcnt path)
        int t = threadIdx.x;
        unsigned long gb = (unsigned long)(uintptr_t)WgP + (unsigned long)(t * 16);
        unsigned      lb = (unsigned)(uintptr_t)(&sw[0]) + (unsigned)(t * 16);
#pragma unroll
        for (int i = 0; i < 8; i++) {
            unsigned long ga = gb + (unsigned long)(i * 4096);
            unsigned      la = lb + (unsigned)(i * 4096);
            asm volatile("global_load_async_to_lds_b128 %0, %1, off"
                         :: "v"(la), "v"(ga) : "memory");
        }
        asm volatile("s_wait_asynccnt 0x0" ::: "memory");
    }
    __syncthreads();

    int wid  = threadIdx.x >> 5;
    int lane = threadIdx.x & 31;
    int w    = blockIdx.x * 8 + wid;        // 8000 waves = 4000 mtiles x 2 halves
    int mt = w >> 1, nh = w & 1;
    int m0 = mt * 16;

    // A fragments for K=0..63 (ISA 7.12.2 16-bit A layout), kept in registers
    int arow    = m0 + (lane & 15);
    int kb_half = (lane < 16) ? 0 : 8;
    const _Float16* ap = Xh + arow * HID + kb_half;
    v8h lo0 = *(const v8h*)ap;
    v8h hi0 = *(const v8h*)(ap + 16);
    v8h lo1 = *(const v8h*)(ap + 32);
    v8h hi1 = *(const v8h*)(ap + 48);
    v16h a0, a1;
#pragma unroll
    for (int i = 0; i < 8; i++) {
        a0[i] = lo0[i]; a0[i + 8] = hi0[i];
        a1[i] = lo1[i]; a1[i + 8] = hi1[i];
    }

    int crow = m0 + (lane >> 4) * 8;
#pragma unroll
    for (int j = 0; j < 8; j++) {
        int nt = nh * 8 + j;
        const _Float16* wp = &sw[((nt * 2) * 32 + lane) * 16];
        v16h b0 = *(const v16h*)wp;                 // ds_load path
        v16h b1 = *(const v16h*)(wp + 32 * 16);
        v8f c = {};
        c = __builtin_amdgcn_wmma_f32_16x16x32_f16(false, a0, false, b0, (short)0, c, false, false);
        c = __builtin_amdgcn_wmma_f32_16x16x32_f16(false, a1, false, b1, (short)0, c, false, false);
        _Float16* hp = H + (size_t)crow * HC + nt * 16 + (lane & 15);
#pragma unroll
        for (int v = 0; v < 8; v++) hp[(size_t)v * HC] = (_Float16)c[v];
    }
}

// ---------------- per-node attention terms ls/ld ----------------

__global__ void k_attn_logits(const _Float16* __restrict__ H, const float* __restrict__ asrc,
                              const float* __restrict__ adst, float* __restrict__ LS,
                              float* __restrict__ LD) {
    int i = blockIdx.x * blockDim.x + threadIdx.x;   // ROWS*HEADS
    if (i >= ROWS * HEADS) return;
    int h  = i & 3;
    int gn = i >> 2;
    const _Float16* hp = H + (size_t)gn * HC + h * HID;
    const float* as = asrc + h * HID;
    const float* ad = adst + h * HID;
    float s = 0.f, d = 0.f;
#pragma unroll
    for (int c2 = 0; c2 < HID; c2++) {
        float hv = (float)hp[c2];
        s += hv * as[c2];
        d += hv * ad[c2];
    }
    LS[i] = s;
    LD[i] = d;
}

// ---------------- softmax-aggregate + mean heads + bias + ELU ----------------
// one wave per (graph,node); lane owns channels c=lane and c=lane+32.
// Fixed CSR order -> deterministic sums. Writes next-layer Xh in place.

__global__ void __launch_bounds__(256) k_aggregate(const _Float16* __restrict__ H,
                                                   const float* __restrict__ LS,
                                                   const float* __restrict__ LD,
                                                   const int* __restrict__ start,
                                                   const int* __restrict__ csrc,
                                                   const float* __restrict__ bgl,
                                                   _Float16* __restrict__ Xh) {
    int wid  = threadIdx.x >> 5;
    int lane = threadIdx.x & 31;
    int idx  = blockIdx.x * 8 + wid;        // 64000 (graph,node) pairs
    int g = idx / N_, n = idx % N_;
    int gb = g * N_;
    int s0 = start[n], s1 = start[n + 1];

    float o0 = 0.f, o1 = 0.f;
    for (int h = 0; h < HEADS; h++) {
        float ldv = LD[(gb + n) * HEADS + h];
        float m = -1e30f;
        for (int e = s0; e < s1; e++) {                 // pass 1: segment max
            int src = csrc[e];
            float v = LS[(gb + src) * HEADS + h] + ldv;
            v = (v > 0.f) ? v : 0.2f * v;
            m = fmaxf(m, v);
        }
        float den = 0.f, a0 = 0.f, a1 = 0.f;
        for (int e = s0; e < s1; e++) {                 // pass 2: exp-sum + weighted acc
            int src = csrc[e];
            float v = LS[(gb + src) * HEADS + h] + ldv;
            v = (v > 0.f) ? v : 0.2f * v;
            float ex = __expf(v - m);
            den += ex;
            const _Float16* hp = H + (size_t)(gb + src) * HC + h * HID;
            a0 += ex * (float)hp[lane];
            a1 += ex * (float)hp[lane + 32];
        }
        float inv = 1.f / (den + 1e-16f);
        o0 += a0 * inv;
        o1 += a1 * inv;
    }
    o0 = 0.25f * o0 + bgl[lane];
    o1 = 0.25f * o1 + bgl[lane + 32];
    o0 = (o0 > 0.f) ? o0 : (__expf(o0) - 1.f);          // ELU
    o1 = (o1 > 0.f) ? o1 : (__expf(o1) - 1.f);
    size_t ob = (size_t)(gb + n) * HID;
    Xh[ob + lane]      = (_Float16)o0;
    Xh[ob + lane + 32] = (_Float16)o1;
}

// ---------------- GRU input GEMM (all timesteps at once, K-split WMMA) --------
// [64 x 64000] @ [64000 x 192]; one wave per (kchunk, ntile) computes all 4
// M-tiles -> W_ih streamed exactly once. 100x12 = 1200 waves.

__global__ void __launch_bounds__(256) k_gemm_gi(const _Float16* __restrict__ Xh,
                                                 const _Float16* __restrict__ Wih,
                                                 float* __restrict__ Gpart) {
    int wid  = threadIdx.x >> 5;
    int lane = threadIdx.x & 31;
    int w = blockIdx.x * 8 + wid;           // 1200 waves
    int kchunk = w / 12, nt = w % 12;
    int col = nt * 16 + (lane & 15);
    const _Float16* bbase = Wih + (size_t)col * KGRU + ((lane < 16) ? 0 : 16);

    const _Float16* abase[4];
#pragma unroll
    for (int mt = 0; mt < 4; mt++) {
        int r  = mt * 16 + (lane & 15);     // GI row = t*4+b
        int gR = (r & 3) * 16 + (r >> 2);   // graph index g = b*16+t
        abase[mt] = Xh + (size_t)gR * KGRU + ((lane < 16) ? 0 : 16);
    }

    v8f c[4] = {v8f{}, v8f{}, v8f{}, v8f{}};
    int kb0 = kchunk * (KCW * 32);
    for (int kk = 0; kk < KCW; kk++) {
        int kb = kb0 + kk * 32;
        __builtin_prefetch(bbase + kb + 256, 0, 0);     // global_prefetch_b8
        v16h b = *(const v16h*)(bbase + kb);
#pragma unroll
        for (int mt = 0; mt < 4; mt++) {
            v8h lo = *(const v8h*)(abase[mt] + kb);
            v8h hi = *(const v8h*)(abase[mt] + kb + 8);
            v16h a;
#pragma unroll
            for (int i = 0; i < 8; i++) { a[i] = lo[i]; a[i + 8] = hi[i]; }
            c[mt] = __builtin_amdgcn_wmma_f32_16x16x32_f16(false, a, false, b, (short)0,
                                                           c[mt], false, false);
        }
    }
#pragma unroll
    for (int mt = 0; mt < 4; mt++) {
        int rr = mt * 16 + (lane >> 4) * 8;
        float* gp = Gpart + ((size_t)kchunk * G_ + rr) * GRUO + col;
#pragma unroll
        for (int v = 0; v < 8; v++) gp[(size_t)v * GRUO] = c[mt][v];
    }
}

__global__ void k_gi_reduce(const float* __restrict__ Gpart, const float* __restrict__ bih,
                            float* __restrict__ GI) {
    int i = blockIdx.x * blockDim.x + threadIdx.x;
    if (i >= G_ * GRUO) return;
    int r = i / GRUO, j = i % GRUO;
    float s = bih[j];
    for (int ch = 0; ch < KCH; ch++)                     // fixed order: deterministic
        s += Gpart[((size_t)ch * G_ + r) * GRUO + j];
    GI[i] = s;
}

// ---------------- sequential GRU scan (tiny; single block) ----------------

__global__ void __launch_bounds__(768) k_gru(const float* __restrict__ GI,
                                             const float* __restrict__ Whh,
                                             const float* __restrict__ bhh,
                                             float* __restrict__ hT) {
    __shared__ float hs[B_ * HID];
    __shared__ float ghs[B_ * GRUO];
    int tid = threadIdx.x;
    if (tid < B_ * HID) hs[tid] = 0.f;
    __syncthreads();
    int b = tid / GRUO, j = tid % GRUO;
    for (int t = 0; t < T_; t++) {
        float gh = bhh[j];
        const float* hp = hs + b * HID;
        const float* wp = Whh + j * HID;
#pragma unroll
        for (int k = 0; k < HID; k++) gh += hp[k] * wp[k];
        ghs[b * GRUO + j] = gh;
        __syncthreads();
        float hnew = 0.f;
        if (tid < B_ * HID) {
            int bb = tid / HID, jj = tid % HID;
            const float* gi = GI + (t * B_ + bb) * GRUO;
            const float* g2 = ghs + bb * GRUO;
            float r  = 1.f / (1.f + __expf(-(gi[jj]           + g2[jj])));
            float z  = 1.f / (1.f + __expf(-(gi[HID + jj]     + g2[HID + jj])));
            float nn = tanhf(gi[2 * HID + jj] + r * g2[2 * HID + jj]);
            hnew = (1.f - z) * nn + z * hs[tid];
        }
        __syncthreads();
        if (tid < B_ * HID) hs[tid] = hnew;
        __syncthreads();
    }
    if (tid < B_ * HID) hT[tid] = hs[tid];
}

// ---------------- output head ----------------

__global__ void k_head(const float* __restrict__ hT, const float* __restrict__ Wfc,
                       const float* __restrict__ bfc, float* __restrict__ out) {
    int i = blockIdx.x * blockDim.x + threadIdx.x;
    if (i >= B_ * N_) return;
    int b = i / N_, n = i % N_;
    float s = bfc[n];
    const float* hp = hT + b * HID;
#pragma unroll
    for (int c = 0; c < HID; c++) s += hp[c] * Wfc[c * N_ + n];
    out[i] = s;
}

// ---------------- launcher ----------------

extern "C" void kernel_launch(void* const* d_in, const int* in_sizes, int n_in,
                              void* d_out, int out_size, void* d_ws, size_t ws_size,
                              hipStream_t stream) {
    (void)in_sizes; (void)n_in; (void)out_size; (void)ws_size;
    const float* x_seq = (const float*)d_in[0];
    const int*   eidx  = (const int*)d_in[1];
    // d_in[2] edge_weight: unused by reference GATConv
    const float* Win = (const float*)d_in[3];
    const float* bin = (const float*)d_in[4];
    const float* Wg  = (const float*)d_in[5];
    const float* asrc= (const float*)d_in[6];
    const float* adst= (const float*)d_in[7];
    const float* bg  = (const float*)d_in[8];
    const float* Wih = (const float*)d_in[9];
    const float* Whh = (const float*)d_in[10];
    const float* bih = (const float*)d_in[11];
    const float* bhh = (const float*)d_in[12];
    const float* Wfc = (const float*)d_in[13];
    const float* bfc = (const float*)d_in[14];
    float* out = (float*)d_out;

    char* w = (char*)d_ws;
    auto alloc = [&](size_t bytes) -> char* {
        char* p = w;
        w += (bytes + 255) & ~(size_t)255;
        return p;
    };
    _Float16* Xh    = (_Float16*)alloc((size_t)ROWS * HID * 2);
    _Float16* H     = (_Float16*)alloc((size_t)ROWS * HC * 2);
    float*    LS    = (float*)alloc((size_t)ROWS * HEADS * 4);
    float*    LD    = (float*)alloc((size_t)ROWS * HEADS * 4);
    _Float16* WgP   = (_Float16*)alloc((size_t)LAY * 16 * 2 * 32 * 16 * 2);
    _Float16* WihH  = (_Float16*)alloc((size_t)GRUO * KGRU * 2);
    float*    Gpart = (float*)alloc((size_t)KCH * G_ * GRUO * 4);
    float*    GI    = (float*)alloc((size_t)G_ * GRUO * 4);
    float*    hT    = (float*)alloc((size_t)B_ * HID * 4);
    int*      deg   = (int*)alloc((N_ + 1) * 4);
    int*      start = (int*)alloc((N_ + 1) * 4);
    int*      dstA  = (int*)alloc(EP * 4);
    int*      srcA  = (int*)alloc(EP * 4);
    int*      csrc  = (int*)alloc(EP * 4);

    // weight prep (recomputed every call: deterministic, capture-safe)
    k_pack_wg<<<(LAY * 16 * 2 * 32 * 16 + 255) / 256, 256, 0, stream>>>(Wg, WgP);
    k_conv_f16<<<2048, 256, 0, stream>>>(Wih, WihH, GRUO * KGRU);

    // deterministic CSR
    k_zero_int<<<(N_ + 1 + 255) / 256, 256, 0, stream>>>(deg, N_ + 1);
    k_edges<<<(EP + 255) / 256, 256, 0, stream>>>(eidx, dstA, srcA, deg);
    k_scan<<<1, 32, 0, stream>>>(deg, start);
    k_scatter<<<(EP + 255) / 256, 256, 0, stream>>>(dstA, srcA, start, csrc);

    // input projection
    k_infc<<<(ROWS * HID + 255) / 256, 256, 0, stream>>>(x_seq, Win, bin, Xh);

    // 5 GAT layers
    for (int l = 0; l < LAY; l++) {
        k_gemm_gat<<<1000, 256, 0, stream>>>(Xh, WgP + (size_t)l * 16 * 2 * 32 * 16, H);
        k_attn_logits<<<(ROWS * HEADS + 255) / 256, 256, 0, stream>>>(
            H, asrc + l * HEADS * HID, adst + l * HEADS * HID, LS, LD);
        k_aggregate<<<8000, 256, 0, stream>>>(H, LS, LD, start, csrc, bg + l * HID, Xh);
    }

    // GRU: batched input GEMM (WMMA, K-split) + deterministic reduce + scan
    k_gemm_gi<<<150, 256, 0, stream>>>(Xh, WihH, Gpart);
    k_gi_reduce<<<(G_ * GRUO + 255) / 256, 256, 0, stream>>>(Gpart, bih, GI);
    k_gru<<<1, 768, 0, stream>>>(GI, Whh, bhh, hT);

    // head
    k_head<<<(B_ * N_ + 255) / 256, 256, 0, stream>>>(hT, Wfc, bfc, out);
}
